// WikiGraphSAGE_3246995276182
// MI455X (gfx1250) — matrix operand order repaired and checked by reading, not compiled, not run on hardware
//
#include <hip/hip_runtime.h>

#define DH 128
#define DIN 300
#define KP_EMB 320
#define DOUT 64
#define GNUM 8

typedef __attribute__((ext_vector_type(16))) __bf16 v16bf;
typedef __attribute__((ext_vector_type(8)))  float  v8f;

union FragBF { v16bf v; unsigned int u[8]; };

__device__ __forceinline__ unsigned int f2bf_pack(float lo, float hi) {
  __bf16 a = (__bf16)lo;
  __bf16 b = (__bf16)hi;
  unsigned short ua = __builtin_bit_cast(unsigned short, a);
  unsigned short ub = __builtin_bit_cast(unsigned short, b);
  return (unsigned int)ua | ((unsigned int)ub << 16);
}

__device__ __forceinline__ float4 ld4(const float* p) { return *(const float4*)p; }

// ---- pack W[128][din] (f32) -> per-lane WMMA B-fragment order:
//      out[((t*8 + nt)*32 + lane)*8 + v] = pack(W[n][k0], W[n][k0+1])
//      n = nt*16 + (lane&15), k0 = t*32 + 16*(v>>2) + 8*(lane>>4) + 2*(v&3)  (zero-padded K)
__global__ void pack_wfrag_kernel(const float* __restrict__ W, unsigned int* __restrict__ out,
                                  int din, int kpad) {
  int total = (kpad / 32) * 8 * 32 * 8;
  for (int i = blockIdx.x * blockDim.x + threadIdx.x; i < total; i += gridDim.x * blockDim.x) {
    int v    = i & 7;
    int lane = (i >> 3) & 31;
    int nt   = (i >> 8) & 7;
    int t    = i >> 11;
    int half = lane >> 4, l16 = lane & 15;
    int n  = nt * 16 + l16;
    int k0 = t * 32 + 16 * (v >> 2) + 8 * half + 2 * (v & 3);
    float lo = (k0     < din) ? W[n * din + k0]     : 0.0f;
    float hi = (k0 + 1 < din) ? W[n * din + k0 + 1] : 0.0f;
    out[i] = f2bf_pack(lo, hi);
  }
}

// ---- vectorized zero fill (b128 stores + scalar tail) ----
__global__ void zero_kernel(float* __restrict__ p, int n) {
  int n4 = n >> 2;
  float4* __restrict__ p4 = (float4*)p;
  float4 z = make_float4(0.f, 0.f, 0.f, 0.f);
  for (int i = blockIdx.x * blockDim.x + threadIdx.x; i < n4; i += gridDim.x * blockDim.x)
    p4[i] = z;
  if (blockIdx.x == 0 && (int)threadIdx.x < (n & 3))
    p[n4 * 4 + threadIdx.x] = 0.f;
}

// Build A fragment from 4 aligned float4 loads (one row per lane, contiguous K spans)
__device__ __forceinline__ FragBF make_afrag(const float* __restrict__ base, float scale) {
  float4 q0 = ld4(base);        // K = 8h+0..3
  float4 q1 = ld4(base + 4);    // K = 8h+4..7
  float4 q2 = ld4(base + 16);   // K = 16+8h+0..3
  float4 q3 = ld4(base + 20);   // K = 16+8h+4..7
  FragBF a;
  a.u[0] = f2bf_pack(q0.x * scale, q0.y * scale);
  a.u[1] = f2bf_pack(q0.z * scale, q0.w * scale);
  a.u[2] = f2bf_pack(q1.x * scale, q1.y * scale);
  a.u[3] = f2bf_pack(q1.z * scale, q1.w * scale);
  a.u[4] = f2bf_pack(q2.x * scale, q2.y * scale);
  a.u[5] = f2bf_pack(q2.z * scale, q2.w * scale);
  a.u[6] = f2bf_pack(q3.x * scale, q3.y * scale);
  a.u[7] = f2bf_pack(q3.z * scale, q3.w * scale);
  return a;
}

// ---- h = x @ W_emb^T + b_emb  (bf16 WMMA, f32 accumulate; no LDS, no branches) ----
__global__ void __launch_bounds__(128)
emb_gemm_kernel(const float* __restrict__ x, const unsigned int* __restrict__ WTf,
                const float* __restrict__ bias, float* __restrict__ out, int nrows) {
  const int lane = threadIdx.x & 31;
  const int wave = threadIdx.x >> 5;
  const int half = lane >> 4;
  const int l16  = lane & 15;
  const int tileM = blockIdx.x * 64 + wave * 16;  // 16-row tile owned by this wave
  if (tileM >= nrows) return;                     // wave-uniform exit (no barriers used)
  const bool full = (tileM + 16) <= nrows;        // wave-uniform
  const int  arow = min(tileM + l16, nrows - 1);  // clamped: loads always safe
  const float* __restrict__ xrow = x + (size_t)arow * DIN;

  v8f c[8] = {};

#pragma unroll
  for (int t = 0; t < 9; ++t) {                   // K = 0..287: full 32-wide steps
    FragBF a = make_afrag(xrow + t * 32 + 8 * half, 1.0f);
#pragma unroll
    for (int nt = 0; nt < 8; ++nt) {
      v16bf b = *(const v16bf*)(WTf + (((size_t)t * 8 + nt) * 32 + lane) * 8);
      c[nt] = __builtin_amdgcn_wmma_f32_16x16x32_bf16(false, a.v, false, b, (short)0,
                                                      c[nt], false, false);
    }
  }
  { // t = 9: K = 288..319, data valid only up to 299 (unconditional clamped loads)
    float4 q0 = ld4(xrow + 288 + 8 * half);       // h0: 288-291, h1: 296-299
    float4 q1 = ld4(xrow + 292);                  // 292-295 (safe for all lanes)
    unsigned int u2 = f2bf_pack(q1.x, q1.y);
    unsigned int u3 = f2bf_pack(q1.z, q1.w);
    FragBF a;
    a.u[0] = f2bf_pack(q0.x, q0.y);
    a.u[1] = f2bf_pack(q0.z, q0.w);
    a.u[2] = (half == 0) ? u2 : 0u;               // h1: K=300,301 padded (select, no branch)
    a.u[3] = (half == 0) ? u3 : 0u;               // h1: K=302,303 padded
    a.u[4] = 0u; a.u[5] = 0u; a.u[6] = 0u; a.u[7] = 0u;  // K >= 304 padded
#pragma unroll
    for (int nt = 0; nt < 8; ++nt) {
      v16bf b = *(const v16bf*)(WTf + (((size_t)9 * 8 + nt) * 32 + lane) * 8);
      c[nt] = __builtin_amdgcn_wmma_f32_16x16x32_bf16(false, a.v, false, b, (short)0,
                                                      c[nt], false, false);
    }
  }

  if (full) {
#pragma unroll
    for (int nt = 0; nt < 8; ++nt)
#pragma unroll
      for (int r = 0; r < 8; ++r) {
        int row = tileM + r + 8 * half;
        int col = nt * 16 + l16;
        out[row * DH + col] = c[nt][r] + bias[col];
      }
  } else {
#pragma unroll
    for (int nt = 0; nt < 8; ++nt)
#pragma unroll
      for (int r = 0; r < 8; ++r) {
        int row = tileM + r + 8 * half;
        int col = nt * 16 + l16;
        if (row < nrows) out[row * DH + col] = c[nt][r] + bias[col];
      }
  }
}

// ---- edge scatter: agg[dst] += h[src]; deg[dst] += 1 ----
__global__ void __launch_bounds__(128)
scatter_kernel(const float* __restrict__ h, const int* __restrict__ ei, int E,
               float* __restrict__ agg, float* __restrict__ deg) {
  int t = threadIdx.x;
  for (int e = blockIdx.x; e < E; e += gridDim.x) {
    int src = ei[e];
    int dst = ei[E + e];
    atomicAdd(&agg[dst * DH + t], h[src * DH + t]);
    if (t == 0) atomicAdd(&deg[dst], 1.0f);
  }
}

// ---- h_out = relu( (agg/deg) @ Wl^T + bl + h @ Wr^T )  (no LDS, no branches) ----
__global__ void __launch_bounds__(128)
sage_gemm_kernel(const float* __restrict__ agg, const float* __restrict__ deg,
                 const float* __restrict__ h,
                 const unsigned int* __restrict__ WlTf, const unsigned int* __restrict__ WrTf,
                 const float* __restrict__ bias, float* __restrict__ out, int nrows) {
  const int lane = threadIdx.x & 31;
  const int wave = threadIdx.x >> 5;
  const int half = lane >> 4;
  const int l16  = lane & 15;
  const int tileM = blockIdx.x * 64 + wave * 16;
  if (tileM >= nrows) return;                     // wave-uniform exit
  const bool full = (tileM + 16) <= nrows;
  const int  arow = min(tileM + l16, nrows - 1);
  const float inv = 1.0f / fmaxf(deg[arow], 1.0f);
  const float* __restrict__ grow = agg + (size_t)arow * DH;
  const float* __restrict__ hrow = h   + (size_t)arow * DH;

  v8f c[8] = {};

#pragma unroll
  for (int t = 0; t < DH / 32; ++t) {
    FragBF aa = make_afrag(grow + t * 32 + 8 * half, inv);
    FragBF ah = make_afrag(hrow + t * 32 + 8 * half, 1.0f);
#pragma unroll
    for (int nt = 0; nt < 8; ++nt) {
      v16bf bl = *(const v16bf*)(WlTf + (((size_t)t * 8 + nt) * 32 + lane) * 8);
      c[nt] = __builtin_amdgcn_wmma_f32_16x16x32_bf16(false, aa.v, false, bl, (short)0,
                                                      c[nt], false, false);
    }
#pragma unroll
    for (int nt = 0; nt < 8; ++nt) {
      v16bf br = *(const v16bf*)(WrTf + (((size_t)t * 8 + nt) * 32 + lane) * 8);
      c[nt] = __builtin_amdgcn_wmma_f32_16x16x32_bf16(false, ah.v, false, br, (short)0,
                                                      c[nt], false, false);
    }
  }

  if (full) {
#pragma unroll
    for (int nt = 0; nt < 8; ++nt)
#pragma unroll
      for (int r = 0; r < 8; ++r) {
        int row = tileM + r + 8 * half;
        int col = nt * 16 + l16;
        out[row * DH + col] = fmaxf(c[nt][r] + bias[col], 0.0f);
      }
  } else {
#pragma unroll
    for (int nt = 0; nt < 8; ++nt)
#pragma unroll
      for (int r = 0; r < 8; ++r) {
        int row = tileM + r + 8 * half;
        int col = nt * 16 + l16;
        if (row < nrows) out[row * DH + col] = fmaxf(c[nt][r] + bias[col], 0.0f);
      }
  }
}

// ---- attention scores: s[i] = h[i,:].Watt + b_att (one wave per node, b128 loads) ----
__global__ void __launch_bounds__(128)
att_score_kernel(const float* __restrict__ h, const float* __restrict__ Watt,
                 const float* __restrict__ batt, float* __restrict__ s, int nrows) {
  int lane = threadIdx.x & 31;
  int wave = threadIdx.x >> 5;
  float4 w4 = *(const float4*)(Watt + lane * 4);
  float bb = batt[0];
  for (int node = blockIdx.x * 4 + wave; node < nrows; node += gridDim.x * 4) {
    float4 hv = *(const float4*)(h + (size_t)node * DH + lane * 4);
    float acc = hv.x * w4.x + hv.y * w4.y + hv.z * w4.z + hv.w * w4.w;
#pragma unroll
    for (int off = 16; off > 0; off >>= 1)
      acc += __shfl_xor(acc, off, 32);
    if (lane == 0) s[node] = acc + bb;
  }
}

__device__ __forceinline__ unsigned int enc_f32(float f) {
  unsigned int u = __float_as_uint(f);
  return (u & 0x80000000u) ? ~u : (u | 0x80000000u);
}
__device__ __forceinline__ float dec_f32(unsigned int e) {
  unsigned int u = (e & 0x80000000u) ? (e & 0x7FFFFFFFu) : ~e;
  return __uint_as_float(u);
}

__global__ void reduce_max_kernel(const float* __restrict__ s, int n,
                                  unsigned int* __restrict__ maxenc) {
  __shared__ float red[256];
  float m = -3.402823466e38f;
  for (int i = blockIdx.x * blockDim.x + threadIdx.x; i < n; i += gridDim.x * blockDim.x)
    m = fmaxf(m, s[i]);
  red[threadIdx.x] = m;
  __syncthreads();
  for (int off = 128; off > 0; off >>= 1) {
    if ((int)threadIdx.x < off) red[threadIdx.x] = fmaxf(red[threadIdx.x], red[threadIdx.x + off]);
    __syncthreads();
  }
  if (threadIdx.x == 0) atomicMax(maxenc, enc_f32(red[0]));
}

__global__ void expsum_kernel(const float* __restrict__ s, int n,
                              const unsigned int* __restrict__ maxenc,
                              float* __restrict__ expw, float* __restrict__ sum) {
  __shared__ float red[256];
  float mx = dec_f32(*maxenc);
  float acc = 0.f;
  for (int i = blockIdx.x * blockDim.x + threadIdx.x; i < n; i += gridDim.x * blockDim.x) {
    float e = __expf(s[i] - mx);
    expw[i] = e;
    acc += e;
  }
  red[threadIdx.x] = acc;
  __syncthreads();
  for (int off = 128; off > 0; off >>= 1) {
    if ((int)threadIdx.x < off) red[threadIdx.x] += red[threadIdx.x + off];
    __syncthreads();
  }
  if (threadIdx.x == 0) atomicAdd(sum, red[0]);
}

// ---- pooled[g] += h[i] * expw[i]/sum, via LDS accumulation ----
__global__ void __launch_bounds__(128)
pool_kernel(const float* __restrict__ h, const float* __restrict__ expw,
            const float* __restrict__ sumv, const int* __restrict__ batch,
            float* __restrict__ pooled, int nrows) {
  __shared__ float pl[GNUM * DH];
  int tid = threadIdx.x;
  for (int i = tid; i < GNUM * DH; i += 128) pl[i] = 0.f;
  __syncthreads();
  float inv = 1.0f / sumv[0];
  for (int node = blockIdx.x; node < nrows; node += gridDim.x) {
    int g = batch[node];
    float w = expw[node] * inv;
    atomicAdd(&pl[g * DH + tid], h[node * DH + tid] * w);
  }
  __syncthreads();
  for (int i = tid; i < GNUM * DH; i += 128) atomicAdd(&pooled[i], pl[i]);
}

// ---- out = pooled @ W_out^T + b_out  (8x64, scalar) ----
__global__ void out_kernel(const float* __restrict__ pooled, const float* __restrict__ Wout,
                           const float* __restrict__ bout, float* __restrict__ out) {
  int t = blockIdx.x * blockDim.x + threadIdx.x;
  if (t >= GNUM * DOUT) return;
  int g = t / DOUT, d = t - g * DOUT;
  float acc = bout[d];
  for (int f = 0; f < DH; ++f) acc += pooled[g * DH + f] * Wout[d * DH + f];
  out[t] = acc;
}

extern "C" void kernel_launch(void* const* d_in, const int* in_sizes, int n_in,
                              void* d_out, int out_size, void* d_ws, size_t ws_size,
                              hipStream_t stream) {
  const float* x     = (const float*)d_in[0];
  const int*   ei    = (const int*)d_in[1];
  const int*   batch = (const int*)d_in[2];
  const float* W_emb = (const float*)d_in[3];
  const float* b_emb = (const float*)d_in[4];
  const float* Wl0   = (const float*)d_in[5];
  const float* bl0   = (const float*)d_in[6];
  const float* Wr0   = (const float*)d_in[7];
  const float* Wl1   = (const float*)d_in[8];
  const float* bl1   = (const float*)d_in[9];
  const float* Wr1   = (const float*)d_in[10];
  const float* W_att = (const float*)d_in[11];
  const float* b_att = (const float*)d_in[12];
  const float* W_out = (const float*)d_in[13];
  const float* b_out = (const float*)d_in[14];

  const int N = in_sizes[0] / DIN;
  const int E = in_sizes[1] / 2;

  char* ws = (char*)d_ws;
  size_t off = 0;
  auto alloc = [&](size_t bytes) -> char* {
    char* p = ws + off;
    off += (bytes + 255) & ~(size_t)255;
    return p;
  };
  float* h0      = (float*)alloc((size_t)N * DH * 4);   // emb out / layer1 out
  float* h1      = (float*)alloc((size_t)N * DH * 4);   // layer0 out
  float* agg     = (float*)alloc((size_t)N * DH * 4);
  float* deg     = (float*)alloc((size_t)N * 4);
  float* scores  = (float*)alloc((size_t)N * 4);
  float* expw    = (float*)alloc((size_t)N * 4);
  float* pooled  = (float*)alloc(GNUM * DH * 4);        // 4096 B
  unsigned int* maxenc = (unsigned int*)alloc(256);
  float* sumv    = (float*)alloc(256);
  unsigned int* WembTf = (unsigned int*)alloc((size_t)(KP_EMB / 32) * 2048 * 4);
  unsigned int* Wl0Tf  = (unsigned int*)alloc((size_t)(DH / 32) * 2048 * 4);
  unsigned int* Wr0Tf  = (unsigned int*)alloc((size_t)(DH / 32) * 2048 * 4);
  unsigned int* Wl1Tf  = (unsigned int*)alloc((size_t)(DH / 32) * 2048 * 4);
  unsigned int* Wr1Tf  = (unsigned int*)alloc((size_t)(DH / 32) * 2048 * 4);

  // pack weights into per-lane WMMA B-fragment order (bf16 pairs, K-padded)
  pack_wfrag_kernel<<<40, 512, 0, stream>>>(W_emb, WembTf, DIN, KP_EMB);
  pack_wfrag_kernel<<<16, 512, 0, stream>>>(Wl0, Wl0Tf, DH, DH);
  pack_wfrag_kernel<<<16, 512, 0, stream>>>(Wr0, Wr0Tf, DH, DH);
  pack_wfrag_kernel<<<16, 512, 0, stream>>>(Wl1, Wl1Tf, DH, DH);
  pack_wfrag_kernel<<<16, 512, 0, stream>>>(Wr1, Wr1Tf, DH, DH);

  const int mtiles = (N + 63) / 64;
  emb_gemm_kernel<<<mtiles, 128, 0, stream>>>(x, WembTf, b_emb, h0, N);

  // ---- SAGE layer 0 ----
  zero_kernel<<<2048, 256, 0, stream>>>(agg, N * DH);
  zero_kernel<<<64, 256, 0, stream>>>(deg, N);
  scatter_kernel<<<4096, 128, 0, stream>>>(h0, ei, E, agg, deg);
  sage_gemm_kernel<<<mtiles, 128, 0, stream>>>(agg, deg, h0, Wl0Tf, Wr0Tf, bl0, h1, N);

  // ---- SAGE layer 1 ----
  zero_kernel<<<2048, 256, 0, stream>>>(agg, N * DH);
  zero_kernel<<<64, 256, 0, stream>>>(deg, N);
  scatter_kernel<<<4096, 128, 0, stream>>>(h1, ei, E, agg, deg);
  sage_gemm_kernel<<<mtiles, 128, 0, stream>>>(agg, deg, h1, Wl1Tf, Wr1Tf, bl1, h0, N);

  // ---- attention readout ----
  att_score_kernel<<<1024, 128, 0, stream>>>(h0, W_att, b_att, scores, N);
  // zero pooled (4096 B) + maxenc (256 B) + sumv (256 B): contiguous allocations
  zero_kernel<<<2, 256, 0, stream>>>(pooled, GNUM * DH + 128);
  reduce_max_kernel<<<512, 256, 0, stream>>>(scores, N, maxenc);
  expsum_kernel<<<512, 256, 0, stream>>>(scores, N, maxenc, expw, sumv);
  pool_kernel<<<512, 128, 0, stream>>>(h0, expw, sumv, batch, pooled, N);
  out_kernel<<<2, 256, 0, stream>>>(pooled, W_out, b_out, (float*)d_out);
}
